// BARefiner_10857677325131
// MI455X (gfx1250) — matrix operand order repaired and checked by reading, not compiled, not run on hardware
//
#include <hip/hip_runtime.h>
#include <hip/hip_bf16.h>

// ---------------------------------------------------------------------------
// BARefiner for MI455X (gfx1250, wave32, WMMA).
//
// Structure (per message-passing iteration):
//   per-node WMMA GEMMs:
//     delta_x = MLP_h(s)                    (128->64->32->3, pad 16)
//     pre1    = s @ W_f0[3:131,:] + b_f0    (128->64)   [s_j part of f layer 1]
//     self_e  = MLP_f(concat(-delta_x, s))  (self-loop edge, per node)
//   per-edge (wave-per-node, 16 contiguous edges == one 16-row WMMA tile):
//     t1 = relu(pre1[j] + (x_j-x_i-dx_i) @ W_pos)   (staged to LDS, f16)
//     t2 = relu(t1 @ W_f1 + b)  [16x64 * 64x32, 4 wmma]
//     e  = t2 @ W_f2 + b        [16x32 * 32x128, 8 wmma]
//     agg[i] = max over 16 rows (v8f max + shfl_xor(16)) fused with self_e
//   s += MLP_g(agg) via WMMA GEMMs.
// Heads: WMMA GEMM -> instance-norm(relu) -> ... -> padded final GEMM.
//
// Param pytree flatten assumption (jax sorted dict keys):
//   d_in[3..8]   cls:  W0,b0,W1,b1,W2,b2
//   d_in[9+18*it ...] layers[it]: f(W0,b0,W1,b1,W2,b2) g(...) h(...)
//   d_in[63+6*c ...]  loc[c]: W0,b0,W1,b1,W2,b2
// ---------------------------------------------------------------------------

typedef __attribute__((ext_vector_type(16))) _Float16 v16h;
typedef __attribute__((ext_vector_type(8)))  _Float16 v8h;
typedef __attribute__((ext_vector_type(8)))  float    v8f;

#define WAVE 32
#define DEGC 16

// ---------------------------------------------------------------------------
// A-fragment loader: 16x32 f16 tile, one row per lane (row = lane&15).
// Layout (matches packed B): lanes 0-15 hold K=0..15 (elems 0..15);
// lanes 16-31 hold K=16..31.  p points at row base + k-chunk.
// ---------------------------------------------------------------------------
__device__ __forceinline__ v16h load_a_frag(const _Float16* __restrict__ p, int h) {
    v8h lo = *(const v8h*)(p + h * 16);
    v8h hi = *(const v8h*)(p + h * 16 + 8);
    v16h a;
#pragma unroll
    for (int i = 0; i < 8; ++i) { a[i] = lo[i]; a[i + 8] = hi[i]; }
    return a;
}

// ---------------------------------------------------------------------------
// Weight pack: W [din_total, dout_act] f32 row-major  ->  fragment-native f16.
// Fragment (kbi, nbi) occupies 512 halves; lane l holds 16 contiguous halves:
//   col n = nbi*16 + (l&15),  K = kbi*32 + (l>>4)*16 + e   (B-matrix layout).
// Padded cols (n >= dout_act) are zero.
// ---------------------------------------------------------------------------
__global__ void pack_w_kernel(const float* __restrict__ W, _Float16* __restrict__ Wp,
                              int din, int dout_pad, int dout_act, int ldw, int row_off) {
    int p = blockIdx.x * 256 + threadIdx.x;
    int total = din * dout_pad;
    if (p >= total) return;
    int e   = p & 15;
    int l   = (p >> 4) & 31;
    int blk = p >> 9;
    int nkb = din >> 5;
    int kbi = blk % nkb;
    int nbi = blk / nkb;
    int h   = l >> 4;
    int n   = (nbi << 4) + (l & 15);
    int k   = (kbi << 5) + (h << 4) + e;
    float v = (n < dout_act) ? W[(size_t)(row_off + k) * ldw + n] : 0.0f;
    Wp[p] = (_Float16)v;
}

// ---------------------------------------------------------------------------
// Generic WMMA GEMM:  Y = act(X[N,din]_f16 @ Wpacked + bias)
// 256 threads = 8 waves; each wave owns a 16-row tile; NB 16-col blocks.
// ---------------------------------------------------------------------------
template <int NB>
__global__ void __launch_bounds__(256) wmma_gemm(
    const _Float16* __restrict__ X, int ldx,
    const _Float16* __restrict__ Wp,
    const float* __restrict__ bias, int dbias,
    float* __restrict__ Yf, int ldyf,
    _Float16* __restrict__ Yh, int ldyh,
    int N, int din, int dstore, int relu) {
    int lane = threadIdx.x & 31;
    int wv   = threadIdx.x >> 5;
    int rowbase = (blockIdx.x * 8 + wv) * 16;
    if (rowbase >= N) return;
    int h = lane >> 4, mn = lane & 15;

    v8f acc[NB];
#pragma unroll
    for (int c = 0; c < NB; ++c) {
        int col = c * 16 + mn;
        float bb = (bias && col < dbias) ? bias[col] : 0.0f;
#pragma unroll
        for (int r = 0; r < 8; ++r) acc[c][r] = bb;
    }

    const _Float16* xrow = X + (size_t)(rowbase + mn) * ldx;
    int nkb = din >> 5;
    for (int kbi = 0; kbi < nkb; ++kbi) {
        v16h a = load_a_frag(xrow + (kbi << 5), h);
#pragma unroll
        for (int c = 0; c < NB; ++c) {
            v16h b = *(const v16h*)(Wp + (((size_t)(c * nkb + kbi)) << 9) + lane * 16);
            acc[c] = __builtin_amdgcn_wmma_f32_16x16x32_f16(
                false, a, false, b, (short)0, acc[c], false, false);
        }
    }

#pragma unroll
    for (int c = 0; c < NB; ++c) {
        int col = c * 16 + mn;
        if (col >= dstore) continue;
#pragma unroll
        for (int r = 0; r < 8; ++r) {
            int row = rowbase + r + 8 * h;   // C/D layout: M = r + 8*half
            float v = acc[c][r];
            if (relu) v = v > 0.0f ? v : 0.0f;
            if (Yf) Yf[(size_t)row * ldyf + col] = v;
            if (Yh) Yh[(size_t)row * ldyh + col] = (_Float16)v;
        }
    }
}

// ---------------------------------------------------------------------------
// Edge kernel: one wave per node i. Its 16 random edges are e = i*16+m.
//   t1[m,:] = relu(pre1[j_m] + dx_m @ Wpos)         (LDS, f16, 16x64)
//   t2 = relu(t1 @ Wf1 + b1)                         (16x32, 4 wmma)
//   e  = t2 @ Wf2 + b2                               (16x128, 8 wmma)
//   agg[i,:] = max(rowmax(e), self_e[i,:])           (v8f max + shfl_xor 16)
// ---------------------------------------------------------------------------
__global__ void __launch_bounds__(256) edge_max_kernel(
    const int* __restrict__ dst, const float* __restrict__ pos,
    const float* __restrict__ dxp, const float* __restrict__ pre1,
    const float* __restrict__ Wpos,                   // rows 0..2 of W_f0 [3][64]
    const _Float16* __restrict__ Wp2, const float* __restrict__ b2,
    const _Float16* __restrict__ Wp3, const float* __restrict__ b3,
    const float* __restrict__ self_e, float* __restrict__ agg, int N) {
    __shared__ alignas(16) _Float16 t1[8][16][64];
    __shared__ alignas(16) _Float16 t2[8][16][32];

    int lane = threadIdx.x & 31;
    int wv   = threadIdx.x >> 5;
    int node = blockIdx.x * 8 + wv;
    int i    = node < N ? node : N - 1;   // N % 8 == 0 here; clamp is benign
    int h = lane >> 4, mn = lane & 15;

    // --- stage t1: lane covers edge m = mn, cols h*32 .. h*32+31 ------------
    int j  = dst[(size_t)i * DEGC + mn];
    float d0 = pos[j * 3 + 0] - pos[i * 3 + 0] - dxp[i * 16 + 0];
    float d1 = pos[j * 3 + 1] - pos[i * 3 + 1] - dxp[i * 16 + 1];
    float d2 = pos[j * 3 + 2] - pos[i * 3 + 2] - dxp[i * 16 + 2];
    const float4* prow = (const float4*)(pre1 + (size_t)j * 64);
#pragma unroll
    for (int q = 0; q < 8; ++q) {
        int c4 = h * 8 + q;
        float4 pv = prow[c4];
        float4 w0 = ((const float4*)Wpos)[c4];
        float4 w1 = ((const float4*)(Wpos + 64))[c4];
        float4 w2 = ((const float4*)(Wpos + 128))[c4];
        t1[wv][mn][c4 * 4 + 0] = (_Float16)fmaxf(pv.x + d0 * w0.x + d1 * w1.x + d2 * w2.x, 0.f);
        t1[wv][mn][c4 * 4 + 1] = (_Float16)fmaxf(pv.y + d0 * w0.y + d1 * w1.y + d2 * w2.y, 0.f);
        t1[wv][mn][c4 * 4 + 2] = (_Float16)fmaxf(pv.z + d0 * w0.z + d1 * w1.z + d2 * w2.z, 0.f);
        t1[wv][mn][c4 * 4 + 3] = (_Float16)fmaxf(pv.w + d0 * w0.w + d1 * w1.w + d2 * w2.w, 0.f);
    }
    __syncthreads();

    // --- layer2: [16x64] @ [64x32], relu -> t2 ------------------------------
    v8f a2[2];
#pragma unroll
    for (int c = 0; c < 2; ++c) {
        float bb = b2[c * 16 + mn];
#pragma unroll
        for (int r = 0; r < 8; ++r) a2[c][r] = bb;
    }
#pragma unroll
    for (int kbi = 0; kbi < 2; ++kbi) {
        v16h a = load_a_frag(&t1[wv][mn][kbi * 32], h);
#pragma unroll
        for (int c = 0; c < 2; ++c) {
            v16h b = *(const v16h*)(Wp2 + (((size_t)(c * 2 + kbi)) << 9) + lane * 16);
            a2[c] = __builtin_amdgcn_wmma_f32_16x16x32_f16(
                false, a, false, b, (short)0, a2[c], false, false);
        }
    }
#pragma unroll
    for (int c = 0; c < 2; ++c)
#pragma unroll
        for (int r = 0; r < 8; ++r)
            t2[wv][r + 8 * h][c * 16 + mn] = (_Float16)fmaxf(a2[c][r], 0.f);
    __syncthreads();

    // --- layer3: [16x32] @ [32x128], fused segment-max ----------------------
    v16h a3 = load_a_frag(&t2[wv][mn][0], h);
#pragma unroll
    for (int c = 0; c < 8; ++c) {
        float bb = b3[c * 16 + mn];
        v8f acc;
#pragma unroll
        for (int r = 0; r < 8; ++r) acc[r] = bb;
        v16h b = *(const v16h*)(Wp3 + (((size_t)c) << 9) + lane * 16);
        acc = __builtin_amdgcn_wmma_f32_16x16x32_f16(
            false, a3, false, b, (short)0, acc, false, false);
        float mx = acc[0];
#pragma unroll
        for (int r = 1; r < 8; ++r) mx = fmaxf(mx, acc[r]);
        mx = fmaxf(mx, __shfl_xor(mx, 16, 32));        // other 8 rows
        int col = c * 16 + mn;
        mx = fmaxf(mx, self_e[(size_t)i * 128 + col]); // self-loop edge
        if (h == 0) agg[(size_t)i * 128 + col] = mx;
    }
}

// ---------------------------------------------------------------------------
// Instance norm over last dim (128) + relu, f32 -> f16.  One wave per node.
// ---------------------------------------------------------------------------
__global__ void instnorm_kernel(const float* __restrict__ Yf,
                                _Float16* __restrict__ Yh, int N) {
    int lane = threadIdx.x & 31, wv = threadIdx.x >> 5;
    int i = blockIdx.x * 8 + wv;
    if (i >= N) return;
    float4 v = ((const float4*)(Yf + (size_t)i * 128))[lane];
    float s  = v.x + v.y + v.z + v.w;
    float ss = v.x * v.x + v.y * v.y + v.z * v.z + v.w * v.w;
#pragma unroll
    for (int off = 16; off; off >>= 1) {
        s  += __shfl_xor(s,  off, 32);
        ss += __shfl_xor(ss, off, 32);
    }
    float mu  = s * (1.0f / 128.0f);
    float var = ss * (1.0f / 128.0f) - mu * mu;
    float inv = rsqrtf(var + 1e-5f);
    _Float16* o = Yh + (size_t)i * 128 + lane * 4;
    o[0] = (_Float16)fmaxf((v.x - mu) * inv, 0.f);
    o[1] = (_Float16)fmaxf((v.y - mu) * inv, 0.f);
    o[2] = (_Float16)fmaxf((v.z - mu) * inv, 0.f);
    o[3] = (_Float16)fmaxf((v.w - mu) * inv, 0.f);
}

// ---------------------------------------------------------------------------
// Elementwise helpers
// ---------------------------------------------------------------------------
__global__ void init_s_kernel(const float* __restrict__ in, float* __restrict__ s,
                              _Float16* __restrict__ sh, size_t n) {
    size_t i = (size_t)blockIdx.x * 256 + threadIdx.x;
    if (i < n) { float v = in[i]; s[i] = v; sh[i] = (_Float16)v; }
}
__global__ void cast16_kernel(const float* __restrict__ in, _Float16* __restrict__ out, size_t n) {
    size_t i = (size_t)blockIdx.x * 256 + threadIdx.x;
    if (i < n) out[i] = (_Float16)in[i];
}
__global__ void add_cast_kernel(float* __restrict__ s, const float* __restrict__ g,
                                _Float16* __restrict__ sh, size_t n) {
    size_t i = (size_t)blockIdx.x * 256 + threadIdx.x;
    if (i < n) { float v = s[i] + g[i]; s[i] = v; sh[i] = (_Float16)v; }
}
// self-loop first layer of f: relu(pre1[i] + (-dx_i) @ Wpos) -> f16
__global__ void selfin_kernel(const float* __restrict__ pre1, const float* __restrict__ dxp,
                              const float* __restrict__ Wpos, _Float16* __restrict__ selfh,
                              size_t n) {
    size_t idx = (size_t)blockIdx.x * 256 + threadIdx.x;
    if (idx >= n) return;
    size_t i = idx >> 6;
    int    c = (int)(idx & 63);
    float v = pre1[idx] - (dxp[i * 16 + 0] * Wpos[c] +
                           dxp[i * 16 + 1] * Wpos[64 + c] +
                           dxp[i * 16 + 2] * Wpos[128 + c]);
    selfh[idx] = (_Float16)fmaxf(v, 0.f);
}

// ---------------------------------------------------------------------------
// Host orchestration
// ---------------------------------------------------------------------------
extern "C" void kernel_launch(void* const* d_in, const int* in_sizes, int n_in,
                              void* d_out, int out_size, void* d_ws, size_t ws_size,
                              hipStream_t stream) {
    (void)out_size; (void)ws_size; (void)n_in;
    const float* nf  = (const float*)d_in[0];
    const float* pos = (const float*)d_in[1];
    const int*   ei  = (const int*)d_in[2];
    const int N = in_sizes[1] / 3;
    const int E = in_sizes[2] / 2;
    const int* dst = ei + E;            // row 1; first N*16 entries = random dst
    auto P = [&](int idx) -> const float* { return (const float*)d_in[idx]; };

    // ---- workspace carve ---------------------------------------------------
    char* w = (char*)d_ws; size_t off = 0;
    auto carve = [&](size_t bytes) -> char* {
        char* p = w + off; off = (off + bytes + 255) & ~(size_t)255; return p;
    };
    float*     s    = (float*)carve((size_t)N * 128 * 4);
    _Float16*  sh   = (_Float16*)carve((size_t)N * 128 * 2);
    _Float16*  cbuf = (_Float16*)carve((size_t)N * 128 * 2);
    _Float16*  t64h = cbuf;                       // chain temp [N,64]
    _Float16*  t32h = cbuf + (size_t)N * 64;      // chain temp [N,32]
    _Float16*  ynh  = cbuf;                       // head norm out [N,128] (after iters)
    float*     pre1 = (float*)carve((size_t)N * 64 * 4);
    _Float16*  aggh = (_Float16*)pre1;            // alias: pre1 dead before aggh lives
    float*     e32  = (float*)carve((size_t)N * 128 * 4);  // self_e / gout / head y
    float*     agg  = (float*)carve((size_t)N * 128 * 4);
    float*     dxp  = (float*)carve((size_t)N * 16 * 4);
    _Float16*  selfh = t64h;
    _Float16* Ph0 = (_Float16*)carve(128 * 64 * 2);
    _Float16* Ph1 = (_Float16*)carve(64 * 32 * 2);
    _Float16* Ph2 = (_Float16*)carve(32 * 16 * 2);
    _Float16* Pfs = (_Float16*)carve(128 * 64 * 2);
    _Float16* Pf1 = (_Float16*)carve(64 * 32 * 2);
    _Float16* Pf2 = (_Float16*)carve(32 * 128 * 2);
    _Float16* Pg0 = (_Float16*)carve(128 * 64 * 2);
    _Float16* Pg1 = (_Float16*)carve(64 * 32 * 2);
    _Float16* Pg2 = (_Float16*)carve(32 * 128 * 2);
    _Float16* Pc0 = (_Float16*)carve(128 * 128 * 2);
    _Float16* Pc1 = (_Float16*)carve(128 * 128 * 2);
    _Float16* Pc2 = (_Float16*)carve(128 * 16 * 2);

    const dim3 gb((N + 127) / 128);      // GEMM grid (8 waves x 16 rows / block)
    const dim3 nb8((N + 7) / 8);         // wave-per-node grid
    auto ce = [](size_t n) { return dim3((unsigned)((n + 255) / 256)); };
    auto pack = [&](const float* W, _Float16* Wp, int din, int dpad, int dact,
                    int ldw, int roff) {
        pack_w_kernel<<<ce((size_t)din * dpad), 256, 0, stream>>>(W, Wp, din, dpad, dact, ldw, roff);
    };

    init_s_kernel<<<ce((size_t)N * 128), 256, 0, stream>>>(nf, s, sh, (size_t)N * 128);

    // ---- message-passing iterations ---------------------------------------
    for (int it = 0; it < 3; ++it) {
        const int B = 9 + it * 18;   // f(0..5) g(6..11) h(12..17)
        const float* Wpos = P(B + 0);                     // rows 0..2 of W_f0
        pack(P(B + 12), Ph0, 128, 64, 64, 64, 0);
        pack(P(B + 14), Ph1, 64, 32, 32, 32, 0);
        pack(P(B + 16), Ph2, 32, 16, 3, 3, 0);
        pack(P(B + 0),  Pfs, 128, 64, 64, 64, 3);         // W_f0 rows 3..130
        pack(P(B + 2),  Pf1, 64, 32, 32, 32, 0);
        pack(P(B + 4),  Pf2, 32, 128, 128, 128, 0);
        pack(P(B + 6),  Pg0, 128, 64, 64, 64, 0);
        pack(P(B + 8),  Pg1, 64, 32, 32, 32, 0);
        pack(P(B + 10), Pg2, 32, 128, 128, 128, 0);

        // delta_x = MLP_h(s)  (padded to 16 cols)
        wmma_gemm<4><<<gb, 256, 0, stream>>>(sh, 128, Ph0, P(B + 13), 64,
                                             nullptr, 0, t64h, 64, N, 128, 64, 1);
        wmma_gemm<2><<<gb, 256, 0, stream>>>(t64h, 64, Ph1, P(B + 15), 32,
                                             nullptr, 0, t32h, 32, N, 64, 32, 1);
        wmma_gemm<1><<<gb, 256, 0, stream>>>(t32h, 32, Ph2, P(B + 17), 3,
                                             dxp, 16, nullptr, 0, N, 32, 16, 0);
        // pre1 = s @ W_s + b_f0
        wmma_gemm<4><<<gb, 256, 0, stream>>>(sh, 128, Pfs, P(B + 1), 64,
                                             pre1, 64, nullptr, 0, N, 128, 64, 0);
        // self-loop edge MLP -> self_e
        selfin_kernel<<<ce((size_t)N * 64), 256, 0, stream>>>(pre1, dxp, Wpos, selfh, (size_t)N * 64);
        wmma_gemm<2><<<gb, 256, 0, stream>>>(selfh, 64, Pf1, P(B + 3), 32,
                                             nullptr, 0, t32h, 32, N, 64, 32, 1);
        wmma_gemm<8><<<gb, 256, 0, stream>>>(t32h, 32, Pf2, P(B + 5), 128,
                                             e32, 128, nullptr, 0, N, 32, 128, 0);
        // edge MLP + fused segment-max
        edge_max_kernel<<<nb8, 256, 0, stream>>>(dst, pos, dxp, pre1, Wpos,
                                                 Pf1, P(B + 3), Pf2, P(B + 5),
                                                 e32, agg, N);
        // s += MLP_g(agg)
        cast16_kernel<<<ce((size_t)N * 128), 256, 0, stream>>>(agg, aggh, (size_t)N * 128);
        wmma_gemm<4><<<gb, 256, 0, stream>>>(aggh, 128, Pg0, P(B + 7), 64,
                                             nullptr, 0, t64h, 64, N, 128, 64, 1);
        wmma_gemm<2><<<gb, 256, 0, stream>>>(t64h, 64, Pg1, P(B + 9), 32,
                                             nullptr, 0, t32h, 32, N, 64, 32, 1);
        wmma_gemm<8><<<gb, 256, 0, stream>>>(t32h, 32, Pg2, P(B + 11), 128,
                                             e32, 128, nullptr, 0, N, 32, 128, 0);
        add_cast_kernel<<<ce((size_t)N * 128), 256, 0, stream>>>(s, e32, sh, (size_t)N * 128);
    }

    // ---- heads -------------------------------------------------------------
    float* out = (float*)d_out;
    auto run_head = [&](int base, int dact, float* ydst, int ldy) {
        pack(P(base + 0), Pc0, 128, 128, 128, 128, 0);
        pack(P(base + 2), Pc1, 128, 128, 128, 128, 0);
        pack(P(base + 4), Pc2, 128, 16, dact, dact, 0);
        wmma_gemm<8><<<gb, 256, 0, stream>>>(sh, 128, Pc0, P(base + 1), 128,
                                             e32, 128, nullptr, 0, N, 128, 128, 0);
        instnorm_kernel<<<nb8, 256, 0, stream>>>(e32, ynh, N);
        wmma_gemm<8><<<gb, 256, 0, stream>>>(ynh, 128, Pc1, P(base + 3), 128,
                                             e32, 128, nullptr, 0, N, 128, 128, 0);
        instnorm_kernel<<<nb8, 256, 0, stream>>>(e32, ynh, N);
        wmma_gemm<1><<<gb, 256, 0, stream>>>(ynh, 128, Pc2, P(base + 5), dact,
                                             ydst, ldy, nullptr, 0, N, 128, dact, 0);
    };
    // reg_pred [N,21] first (loc heads, 7 cols each), then cls_pred [N,3]
    for (int c = 0; c < 3; ++c) run_head(63 + c * 6, 7, out + c * 7, 21);
    run_head(3, 3, out + (size_t)N * 21, 3);
}
